// GraphAttentionLayer_78537771975042
// MI455X (gfx1250) — matrix operand order repaired and checked by reading, not compiled, not run on hardware
//
#include <hip/hip_runtime.h>
#include <hip/hip_bf16.h>
#include <stdint.h>

// ---------- CDNA5 WMMA types ----------
typedef __bf16 v16bf __attribute__((ext_vector_type(16)));
typedef float  v8f   __attribute__((ext_vector_type(8)));

union Frag16 {            // 16 bf16 values = 8 VGPRs
    v16bf    v;
    uint32_t u[8];
    uint16_t us[16];
};

#define B_ 8
#define N_ 4096
#define F_ 128
#define C_ 128
#define NEG_INF (-1.0e9f)
#define L2E 1.4426950408889634f

static __device__ inline uint16_t f2bf(float f) {
    union { float f; uint32_t u; } c; c.f = f;
    uint32_t u = c.u;
    return (uint16_t)((u + 0x7FFFu + ((u >> 16) & 1u)) >> 16);
}

// pack two fp32 (given as raw bits) into one bf16 pair: 1 v_perm + 2 v_add
static __device__ inline uint32_t bfpack_rne(uint32_t hi, uint32_t lo) {
    return __builtin_amdgcn_perm(hi + 0x8000u, lo + 0x8000u, 0x07060302u);
}
// truncating pack (fine for probabilities in [0,1]): 1 v_perm
static __device__ inline uint32_t bfpack_tr(float hi, float lo) {
    return __builtin_amdgcn_perm(__float_as_uint(hi), __float_as_uint(lo), 0x07060302u);
}

#if defined(__has_builtin)
#if __has_builtin(__builtin_amdgcn_exp2f)
#define EXP2F(x) __builtin_amdgcn_exp2f(x)
#endif
#if __has_builtin(__builtin_amdgcn_tensor_load_to_lds)
#define HAVE_TDM 1
#endif
#endif
#ifndef EXP2F
#define EXP2F(x) __expf((x) * 0.69314718056f)
#endif

#ifdef HAVE_TDM
typedef uint32_t tdm_g0_t __attribute__((ext_vector_type(4)));
typedef int      tdm_g1_t __attribute__((ext_vector_type(8)));
typedef int      tdm_g2_t __attribute__((ext_vector_type(4)));
typedef int      tdm_g4_t __attribute__((ext_vector_type(8)));
#endif

// =====================================================================
// Kernel 0: W [F,C] fp32 -> Wt [C,F] bf16
// =====================================================================
__global__ void k_wt(const float* __restrict__ W, uint16_t* __restrict__ Wt) {
    int idx = blockIdx.x * blockDim.x + threadIdx.x;   // idx = c*128 + f
    if (idx >= C_ * F_) return;
    int c = idx >> 7, f = idx & 127;
    Wt[idx] = f2bf(W[f * C_ + c]);
}

// =====================================================================
// Kernel 1: dense adjacency A [N,N] fp32 -> bitmask [N, N/32] u32
// =====================================================================
__global__ void k_mask(const float* __restrict__ A, uint32_t* __restrict__ bits) {
    int idx = blockIdx.x * blockDim.x + threadIdx.x;   // idx = row*128 + jt
    if (idx >= N_ * (N_ / 32)) return;
    int row = idx >> 7, jt = idx & 127;
    const float* p = A + (size_t)row * N_ + jt * 32;
    uint32_t w = 0;
#pragma unroll
    for (int k = 0; k < 32; ++k)
        if (p[k] > 0.0f) w |= (1u << k);
    bits[idx] = w;
}

// =====================================================================
// Kernel 2: h = x@W (bf16 WMMA, fp32 acc) -> Ht[b][c][n] bf16, e_src/e_dst
// Block = 256 thr (8 waves), 16 rows/block, wave w owns cols [16w,16w+16)
// =====================================================================
__global__ __launch_bounds__(256) void k_h(
    const float* __restrict__ x, const uint16_t* __restrict__ Wt,
    const float* __restrict__ a_src, const float* __restrict__ a_dst,
    uint16_t* __restrict__ Ht, float* __restrict__ e_src, float* __restrict__ e_dst)
{
    const int b  = blockIdx.x >> 8;
    const int i0 = (blockIdx.x & 255) << 4;
    const int lane = threadIdx.x & 31;
    const int wv   = threadIdx.x >> 5;
    const int h = lane >> 4, M = lane & 15;

    const uint32_t* xrow = (const uint32_t*)(x + ((size_t)(b * N_ + i0 + M)) * F_);

    v8f acc = {0.f,0.f,0.f,0.f,0.f,0.f,0.f,0.f};
#pragma unroll
    for (int k0 = 0; k0 < F_; k0 += 32) {
        // A fragment: two contiguous 8-float runs of x -> bf16 (perm pack)
        Frag16 a;
        const uint4* p0 = (const uint4*)(xrow + k0 + h * 8);
        const uint4* p1 = (const uint4*)(xrow + k0 + 16 + h * 8);
        uint4 q0 = p0[0], q1 = p0[1], q2 = p1[0], q3 = p1[1];
        a.u[0] = bfpack_rne(q0.y, q0.x); a.u[1] = bfpack_rne(q0.w, q0.z);
        a.u[2] = bfpack_rne(q1.y, q1.x); a.u[3] = bfpack_rne(q1.w, q1.z);
        a.u[4] = bfpack_rne(q2.y, q2.x); a.u[5] = bfpack_rne(q2.w, q2.z);
        a.u[6] = bfpack_rne(q3.y, q3.x); a.u[7] = bfpack_rne(q3.w, q3.z);
        // B fragment: Wt[c][k], two 8-bf16 runs at k0+8h and k0+16+8h
        Frag16 bb;
        const uint16_t* wrow = Wt + (size_t)(16 * wv + M) * F_ + k0 + h * 8;
        uint4 w0 = *(const uint4*)wrow, w1 = *(const uint4*)(wrow + 16);
        bb.u[0]=w0.x; bb.u[1]=w0.y; bb.u[2]=w0.z; bb.u[3]=w0.w;
        bb.u[4]=w1.x; bb.u[5]=w1.y; bb.u[6]=w1.z; bb.u[7]=w1.w;
        acc = __builtin_amdgcn_wmma_f32_16x16x32_bf16(false, a.v, false, bb.v,
                                                      (short)0, acc, false, false);
    }

    __shared__ float hl[16 * 128];
#pragma unroll
    for (int r = 0; r < 8; ++r)
        hl[(r + 8 * h) * 128 + 16 * wv + M] = acc[r];
    __syncthreads();

    if (threadIdx.x < 16) {
        const float* hr = hl + threadIdx.x * 128;
        float es = 0.f, ed = 0.f;
#pragma unroll 8
        for (int c = 0; c < C_; ++c) { es += hr[c] * a_src[c]; ed += hr[c] * a_dst[c]; }
        e_src[b * N_ + i0 + threadIdx.x] = es;
        e_dst[b * N_ + i0 + threadIdx.x] = ed;
    }

    {   // store transposed bf16 H: Ht[b][c][j]
        int c = threadIdx.x >> 1, halfj = threadIdx.x & 1;
        union { uint32_t u[4]; uint4 q; } pk;
#pragma unroll
        for (int k = 0; k < 4; ++k) {
            uint32_t lo = __float_as_uint(hl[(halfj * 8 + 2 * k + 0) * 128 + c]);
            uint32_t hi = __float_as_uint(hl[(halfj * 8 + 2 * k + 1) * 128 + c]);
            pk.u[k] = bfpack_rne(hi, lo);
        }
        *(uint4*)(Ht + ((size_t)(b * C_ + c)) * N_ + i0 + halfj * 8) = pk.q;
    }
}

// =====================================================================
// Kernel 3: fused masked softmax + aggregation (flash-style).
// Block = 128 thr (4 waves); each wave owns a 16-row tile.
// =====================================================================
__global__ __launch_bounds__(128) void k_attn(
    const float* __restrict__ e_src_g, const float* __restrict__ e_dst_g,
    const uint32_t* __restrict__ bits, const uint16_t* __restrict__ Ht,
    const float* __restrict__ bias, float* __restrict__ out)
{
    const int b  = blockIdx.x >> 6;
    const int wv = threadIdx.x >> 5;
    const int i0 = ((blockIdx.x & 63) << 6) + (wv << 4);
    const int lane = threadIdx.x & 31;
    const int h = lane >> 4, M = lane & 15;
    const int row = i0 + M;

    __shared__ float eds[N_];
    __shared__ float lshare[64];

    // ---- stage e_dst[b,:] (16 KB) into LDS via Tensor Data Mover ----
#ifdef HAVE_TDM
    if (threadIdx.x < 32) {   // wave 0 issues the DMA (EXEC ignored by TDM)
        uint64_t ga = (uint64_t)(uintptr_t)(e_dst_g + (size_t)b * N_);
        uint32_t la = (uint32_t)(uintptr_t)(&eds[0]);       // low 32 bits = LDS offset
        tdm_g0_t g0 = { 1u,                                  // count=1
                        la,                                  // lds_addr
                        (uint32_t)ga,                        // global_addr[31:0]
                        (uint32_t)((ga >> 32) & 0x01FFFFFFull) | 0x80000000u }; // addr[56:32] | type=2
        tdm_g1_t g1 = { (int)0x00020000u,   // data_size = 2 (4 bytes)
                        (int)0x10000000u,   // tensor_dim0 low16 (4096) << 16
                        (int)0x00010000u,   // tensor_dim1 = 1
                        (int)0x10000000u,   // tile_dim0 = 4096
                        1,                  // tile_dim1 = 1
                        4096,               // tensor_dim0_stride low 32
                        0, 0 };
        tdm_g2_t g2 = {0,0,0,0};
        tdm_g2_t g3 = {0,0,0,0};
        tdm_g4_t g4 = {0,0,0,0,0,0,0,0};
        __builtin_amdgcn_tensor_load_to_lds(g0, g1, g2, g3, g4, 0);
        __builtin_amdgcn_s_wait_tensorcnt(0);
    }
#else
    {
        const float4* src = (const float4*)(e_dst_g + (size_t)b * N_);
        float4* dst = (float4*)eds;
#pragma unroll
        for (int k = 0; k < 8; ++k)
            dst[k * 128 + threadIdx.x] = src[k * 128 + threadIdx.x];
    }
#endif
    __syncthreads();

    const float es  = e_src_g[(size_t)b * N_ + row];
    const float esL = es * L2E;
    const uint32_t* mrow = bits + (size_t)row * (N_ / 32);

    // ---------------- pass A: m = leaky(es + max_{j in nbr} ed_j) ----------------
    float mx = -3.0e38f;
    for (int jt = 0; jt < N_ / 32; ++jt) {
        const uint32_t mw = mrow[jt];
        const int j0 = jt * 32;
        const float4* c1 = (const float4*)(eds + j0 + h * 8);
        const float4* c2 = (const float4*)(eds + j0 + 16 + h * 8);
        float v1[8] = {c1[0].x,c1[0].y,c1[0].z,c1[0].w, c1[1].x,c1[1].y,c1[1].z,c1[1].w};
        float v2[8] = {c2[0].x,c2[0].y,c2[0].z,c2[0].w, c2[1].x,c2[1].y,c2[1].z,c2[1].w};
        const uint32_t b1 = (mw >> (h * 8)) & 0xFFu;
        const uint32_t b2 = (mw >> (16 + h * 8)) & 0xFFu;
#pragma unroll
        for (int i = 0; i < 8; ++i) {
            mx = fmaxf(mx, ((b1 >> i) & 1u) ? v1[i] : -3.0e38f);
            mx = fmaxf(mx, ((b2 >> i) & 1u) ? v2[i] : -3.0e38f);
        }
    }
    mx = fmaxf(mx, __shfl_xor(mx, 16, 32));
    const float um = es + mx;
    const float mL = fmaxf(um, 0.2f * um) * L2E;   // m * log2(e)

    // ---------------- pass B: p = exp2(max(w,0.2w) - mL), acc += P x Ht ----------------
    v8f acc[8];
#pragma unroll
    for (int t = 0; t < 8; ++t) acc[t] = (v8f){0.f,0.f,0.f,0.f,0.f,0.f,0.f,0.f};
    float l = 0.f;

    const uint16_t* hbase = Ht + (size_t)b * C_ * N_ + h * 8;

    for (int jt = 0; jt < N_ / 32; ++jt) {
        const uint32_t mw = mrow[jt];
        const int j0 = jt * 32;
        const float4* c1 = (const float4*)(eds + j0 + h * 8);
        const float4* c2 = (const float4*)(eds + j0 + 16 + h * 8);
        float v1[8] = {c1[0].x,c1[0].y,c1[0].z,c1[0].w, c1[1].x,c1[1].y,c1[1].z,c1[1].w};
        float v2[8] = {c2[0].x,c2[0].y,c2[0].z,c2[0].w, c2[1].x,c2[1].y,c2[1].z,c2[1].w};
        const uint32_t b1 = (mw >> (h * 8)) & 0xFFu;
        const uint32_t b2 = (mw >> (16 + h * 8)) & 0xFFu;

        float pv[16];
#pragma unroll
        for (int i = 0; i < 8; ++i) {
            float w1 = fmaf(v1[i], L2E, esL);                       // u*log2e
            float aa = fmaxf(w1 - mL, fmaf(0.2f, w1, -mL));         // leaky in log2 domain
            aa = ((b1 >> i) & 1u) ? aa : NEG_INF;
            float p = EXP2F(aa); l += p; pv[i] = p;

            float w2 = fmaf(v2[i], L2E, esL);
            float ab = fmaxf(w2 - mL, fmaf(0.2f, w2, -mL));
            ab = ((b2 >> i) & 1u) ? ab : NEG_INF;
            float q = EXP2F(ab); l += q; pv[8 + i] = q;
        }

        Frag16 pf;
#pragma unroll
        for (int v = 0; v < 8; ++v)
            pf.u[v] = bfpack_tr(pv[2 * v + 1], pv[2 * v]);

        const uint16_t* hb = hbase + j0;
#pragma unroll
        for (int t = 0; t < 8; ++t) {
            const uint4* q = (const uint4*)(hb + (size_t)(16 * t + M) * N_);
            uint4 q0 = q[0], q1 = q[2];
            Frag16 bf;
            bf.u[0]=q0.x; bf.u[1]=q0.y; bf.u[2]=q0.z; bf.u[3]=q0.w;
            bf.u[4]=q1.x; bf.u[5]=q1.y; bf.u[6]=q1.z; bf.u[7]=q1.w;
            acc[t] = __builtin_amdgcn_wmma_f32_16x16x32_bf16(false, pf.v, false, bf.v,
                                                             (short)0, acc[t], false, false);
        }
    }

    // ---------------- epilogue: 1/l, bias, LeakyReLU(0.3) ----------------
    l += __shfl_xor(l, 16, 32);
    lshare[wv * 16 + M] = l;
    __syncthreads();

    float inv[8];
#pragma unroll
    for (int r = 0; r < 8; ++r)
        inv[r] = 1.0f / lshare[wv * 16 + 8 * h + r];

#pragma unroll
    for (int t = 0; t < 8; ++t) {
        const float bc = bias[16 * t + M];
#pragma unroll
        for (int r = 0; r < 8; ++r) {
            float o = acc[t][r] * inv[r] + bc;
            o = o > 0.f ? o : 0.3f * o;
            out[((size_t)(b * N_ + i0 + 8 * h + r)) * C_ + 16 * t + M] = o;
        }
    }
}

// =====================================================================
extern "C" void kernel_launch(void* const* d_in, const int* in_sizes, int n_in,
                              void* d_out, int out_size, void* d_ws, size_t ws_size,
                              hipStream_t stream) {
    const float* x     = (const float*)d_in[0];
    const float* A     = (const float*)d_in[1];
    const float* W     = (const float*)d_in[2];
    const float* bias  = (const float*)d_in[3];
    const float* a_src = (const float*)d_in[4];
    const float* a_dst = (const float*)d_in[5];
    float* out = (float*)d_out;

    char* ws = (char*)d_ws;
    uint16_t* Wt    = (uint16_t*)ws;                    ws += (size_t)C_ * F_ * 2;
    uint32_t* bits  = (uint32_t*)ws;                    ws += (size_t)N_ * (N_/32) * 4;
    uint16_t* Ht    = (uint16_t*)ws;                    ws += (size_t)B_ * C_ * N_ * 2;
    float*    e_src = (float*)ws;                       ws += (size_t)B_ * N_ * 4;
    float*    e_dst = (float*)ws;

    k_wt  <<<(C_ * F_ + 255) / 256, 256, 0, stream>>>(W, Wt);
    k_mask<<<(N_ * (N_/32) + 255) / 256, 256, 0, stream>>>(A, bits);
    k_h   <<<B_ * (N_ / 16), 256, 0, stream>>>(x, Wt, a_src, a_dst, Ht, e_src, e_dst);
    k_attn<<<B_ * (N_ / 64), 128, 0, stream>>>(e_src, e_dst, bits, Ht, bias, out);
}
